// DependencyParseModel_63763084477102
// MI455X (gfx1250) — matrix operand order repaired and checked by reading, not compiled, not run on hardware
//
#include <hip/hip_runtime.h>
#include <hip/hip_bf16.h>

// ---------------------------------------------------------------------------
// Types for CDNA5 WMMA (gfx1250, wave32)
// ---------------------------------------------------------------------------
typedef __attribute__((ext_vector_type(16))) __bf16 v16bf;
typedef __attribute__((ext_vector_type(8)))  __bf16 v8bf;
typedef __attribute__((ext_vector_type(8)))  float  v8f;

#define T_SEQ 256
#define H_DIM 400
#define G_DIM 1600   // 4*H
#define K400P 416    // 400 padded to multiple of 32
#define MLPH  400

// ---------------------------------------------------------------------------
// f32 -> bf16 conversion with zero padding; rows = gridDim.x
// ---------------------------------------------------------------------------
__global__ void f32_to_bf16_pad_kernel(const float* __restrict__ src, int srcStride,
                                       int colOff, __bf16* __restrict__ dst,
                                       int dstStride, int cols) {
    int r = blockIdx.x;
    for (int c = threadIdx.x; c < dstStride; c += blockDim.x) {
        float v = (c < cols) ? src[(size_t)r * srcStride + colOff + c] : 0.0f;
        dst[(size_t)r * dstStride + c] = (__bf16)v;
    }
}

// ---------------------------------------------------------------------------
// Transpose f32 (R x C) -> (C x R); writes coalesced
// ---------------------------------------------------------------------------
__global__ void transpose_f32_kernel(const float* __restrict__ in, float* __restrict__ out,
                                     int R, int C) {
    int idx = blockIdx.x * blockDim.x + threadIdx.x;
    if (idx < R * C) {
        int c = idx / R;
        int r = idx - c * R;
        out[idx] = in[(size_t)r * C + c];   // out[c*R + r] = in[r][c]
    }
}

// ---------------------------------------------------------------------------
// bias combine: o = a + b
// ---------------------------------------------------------------------------
__global__ void bias_add_kernel(const float* __restrict__ a, const float* __restrict__ b,
                                float* __restrict__ o, int n) {
    int i = blockIdx.x * blockDim.x + threadIdx.x;
    if (i < n) o[i] = a[i] + b[i];
}

// ---------------------------------------------------------------------------
// Embedding gather -> bf16 x (256 x 416, cols 400..415 zero)
// ---------------------------------------------------------------------------
__global__ void embed_kernel(const int* __restrict__ words, const int* __restrict__ tags,
                             const float* __restrict__ word_emb,
                             const float* __restrict__ tag_emb,
                             __bf16* __restrict__ xbf) {
    int t = blockIdx.x;
    int c = threadIdx.x;
    if (c >= K400P) return;
    float v = 0.0f;
    if (c < 300)       v = word_emb[(size_t)words[t] * 300 + c];
    else if (c < 400)  v = tag_emb[(size_t)tags[t] * 100 + (c - 300)];
    xbf[(size_t)t * K400P + c] = (__bf16)v;
}

// ---------------------------------------------------------------------------
// Strip-mined, ping-pong double-buffered WMMA GEMM:
//   C(M x N) = A(M x K) @ B(N x K)^T [+ bias[n]]
// A, B bf16 row-major; acc f32. K % 32 == 0, M % 16 == 0, Ntiles % NSTRIP == 0.
// Each wave owns a 16 x (16*NSTRIP) output strip. Two explicit fragment banks:
// while the matrix pipe consumes bank X, VMEM fills bank Y — no register
// rotation copies, no WMMA->VALU WAR hazards, partial s_wait_loadcnt.
// gridDim.x * 8 == (M/16) * (Ntiles/NSTRIP). Optional transposed store.
// ---------------------------------------------------------------------------
template <int NSTRIP>
struct WmmaFrags {
    v16bf av;
    v16bf bv[NSTRIP];
};

template <int NSTRIP>
__device__ __forceinline__ void load_frags(WmmaFrags<NSTRIP>& f,
                                           const __bf16* __restrict__ aptr,
                                           const __bf16* const* bptr, int k) {
    v8bf a0 = *(const v8bf*)(aptr + k);
    v8bf a1 = *(const v8bf*)(aptr + k + 16);
    f.av = __builtin_shufflevector(a0, a1, 0,1,2,3,4,5,6,7,8,9,10,11,12,13,14,15);
#pragma unroll
    for (int i = 0; i < NSTRIP; ++i) {
        v8bf b0 = *(const v8bf*)(bptr[i] + k);
        v8bf b1 = *(const v8bf*)(bptr[i] + k + 16);
        f.bv[i] = __builtin_shufflevector(b0, b1, 0,1,2,3,4,5,6,7,8,9,10,11,12,13,14,15);
    }
}

template <int NSTRIP>
__device__ __forceinline__ void wmma_consume(v8f (&acc)[NSTRIP],
                                             const WmmaFrags<NSTRIP>& f) {
#pragma unroll
    for (int i = 0; i < NSTRIP; ++i)
        acc[i] = __builtin_amdgcn_wmma_f32_16x16x32_bf16(
            false, f.av, false, f.bv[i], (short)0, acc[i], false, false);
}

template <int NSTRIP>
__global__ __launch_bounds__(256) void wmma_gemm_bt_kernel(
    const __bf16* __restrict__ A, int lda,
    const __bf16* __restrict__ B, int ldb,
    float* __restrict__ C, int ldc,
    const float* __restrict__ bias,
    int Ntiles, int K, int transC)
{
    const int lane = threadIdx.x & 31;
    const int wave = threadIdx.x >> 5;
    const int strip = blockIdx.x * 8 + wave;
    const int strips_per_row = Ntiles / NSTRIP;
    const int tm  = strip / strips_per_row;
    const int tn0 = (strip - tm * strips_per_row) * NSTRIP;
    const int l15 = lane & 15;
    const int hi  = lane >> 4;

    // A fragment: lane holds row (tm*16+l15); K chunks [base..base+7] and
    // [base+16..base+23], base = hi ? 8 : 0  (ISA 16-bit A 16x32 layout)
    const __bf16* aptr = A + (size_t)(tm * 16 + l15) * lda + (hi ? 8 : 0);

    // B fragments: lane holds col ((tn0+i)*16+l15); K chunks [base..base+7] and [base+16..base+23], base = hi ? 8 : 0
    const __bf16* bptr[NSTRIP];
#pragma unroll
    for (int i = 0; i < NSTRIP; ++i)
        bptr[i] = B + (size_t)((tn0 + i) * 16 + l15) * ldb + (hi ? 8 : 0);

    v8f acc[NSTRIP];
#pragma unroll
    for (int i = 0; i < NSTRIP; ++i) acc[i] = (v8f){};

    WmmaFrags<NSTRIP> f0, f1;   // ping-pong banks

    load_frags<NSTRIP>(f0, aptr, bptr, 0);

    int k = 32;
    for (; k + 64 <= K; k += 64) {
        load_frags<NSTRIP>(f1, aptr, bptr, k);        // fill bank1
        wmma_consume<NSTRIP>(acc, f0);                // chew bank0
        load_frags<NSTRIP>(f0, aptr, bptr, k + 32);   // fill bank0
        wmma_consume<NSTRIP>(acc, f1);                // chew bank1
    }
    if (k < K) {   // even number of k-steps
        load_frags<NSTRIP>(f1, aptr, bptr, k);
        wmma_consume<NSTRIP>(acc, f0);
        wmma_consume<NSTRIP>(acc, f1);
    } else {       // odd number of k-steps (K=416 -> 13, K=800 -> 25)
        wmma_consume<NSTRIP>(acc, f0);
    }

    // C/D layout (32-bit, 16x16): lanes 0-15 N=lane, VGPR r -> M=r; lanes 16-31 -> M=8+r
    const int mb = tm * 16 + (hi ? 8 : 0);
#pragma unroll
    for (int i = 0; i < NSTRIP; ++i) {
        const int n = (tn0 + i) * 16 + l15;
        const float bv = bias ? bias[n] : 0.0f;
#pragma unroll
        for (int r = 0; r < 8; ++r) {
            float v = acc[i][r] + bv;
            if (transC) C[(size_t)n * ldc + (mb + r)] = v;
            else        C[(size_t)(mb + r) * ldc + n] = v;
        }
    }
}

// ---------------------------------------------------------------------------
// Persistent LSTM recurrence: one block per direction (gridDim.x == 2).
// Xproj[dir][t][1600] already holds x@Wih.T + bih + bhh.
// WhhT[dir] is (400 x 1600) so inner-k loads are lane-coalesced.
// 800 threads: thread j owns gate rows 2j, 2j+1.
// Writes h_t (bf16) into xcat[t][dir*400 + j].
// ---------------------------------------------------------------------------
__device__ __forceinline__ float sigmoidf_(float x) { return 1.0f / (1.0f + __expf(-x)); }

__global__ __launch_bounds__(800) void lstm_layer_kernel(
    const float* __restrict__ Xproj,   // [2][256][1600]
    const float* __restrict__ WhhT,    // [2][400][1600]
    const float* __restrict__ h0,      // [2][400] (this layer's slice)
    const float* __restrict__ c0,      // [2][400]
    __bf16* __restrict__ xcat, int xld)
{
    const int dir = blockIdx.x;
    const int j   = threadIdx.x;       // 0..799
    const float* Xp = Xproj + (size_t)dir * T_SEQ * G_DIM;
    const float* W  = WhhT  + (size_t)dir * H_DIM * G_DIM;

    __shared__ __align__(16) float h_s[H_DIM];
    __shared__ float c_s[H_DIM];
    __shared__ float g_s[G_DIM];

    if (j < H_DIM) {
        h_s[j] = h0[dir * H_DIM + j];
        c_s[j] = c0[dir * H_DIM + j];
    }
    __syncthreads();

    const int r0 = 2 * j;
    const float* wp = W + r0;

    for (int s = 0; s < T_SEQ; ++s) {
        const int t = dir ? (T_SEQ - 1 - s) : s;
        // warm next timestep's gate-bias row (global_prefetch_b8)
        if (s + 1 < T_SEQ) {
            const int t2 = dir ? (T_SEQ - 2 - s) : (s + 1);
            __builtin_prefetch(Xp + (size_t)t2 * G_DIM + r0, 0, 0);
        }
        const float* xp = Xp + (size_t)t * G_DIM;
        float acc0 = xp[r0];
        float acc1 = xp[r0 + 1];
        for (int k = 0; k < H_DIM; k += 4) {
            float4 hk = *(const float4*)(h_s + k);
            float2 w0 = *(const float2*)(wp + (size_t)(k)     * G_DIM);
            float2 w1 = *(const float2*)(wp + (size_t)(k + 1) * G_DIM);
            float2 w2 = *(const float2*)(wp + (size_t)(k + 2) * G_DIM);
            float2 w3 = *(const float2*)(wp + (size_t)(k + 3) * G_DIM);
            acc0 = fmaf(hk.x, w0.x, acc0); acc1 = fmaf(hk.x, w0.y, acc1);
            acc0 = fmaf(hk.y, w1.x, acc0); acc1 = fmaf(hk.y, w1.y, acc1);
            acc0 = fmaf(hk.z, w2.x, acc0); acc1 = fmaf(hk.z, w2.y, acc1);
            acc0 = fmaf(hk.w, w3.x, acc0); acc1 = fmaf(hk.w, w3.y, acc1);
        }
        g_s[r0] = acc0;
        g_s[r0 + 1] = acc1;
        __syncthreads();
        if (j < H_DIM) {
            float ig = sigmoidf_(g_s[j]);
            float fg = sigmoidf_(g_s[H_DIM + j]);
            float gg = tanhf(g_s[2 * H_DIM + j]);
            float og = sigmoidf_(g_s[3 * H_DIM + j]);
            float c  = fg * c_s[j] + ig * gg;
            float h  = og * tanhf(c);
            c_s[j] = c;
            h_s[j] = h;
            xcat[(size_t)t * xld + dir * H_DIM + j] = (__bf16)h;
        }
        __syncthreads();
    }
}

// ---------------------------------------------------------------------------
// Fused scorer: out[i][j] = (i==j) ? 0 : b2 + sum_k tanh(pi[i][k]+pj[j][k]+b1[k])*W2[k]
// pjT is (400 x 256) so the k-loop load is lane-coalesced. pi row / b1 / W2 in LDS.
// ---------------------------------------------------------------------------
__global__ __launch_bounds__(256) void score_kernel(
    const float* __restrict__ pi,   // [256][400]
    const float* __restrict__ pjT,  // [400][256]
    const float* __restrict__ b1,   // [400]
    const float* __restrict__ W2,   // [400]
    const float* __restrict__ b2,   // [1]
    float* __restrict__ out)        // [256][256]
{
    __shared__ float s_pb[MLPH];
    __shared__ float s_w2[MLPH];
    const int i = blockIdx.x;
    const int j = threadIdx.x;
    for (int k = j; k < MLPH; k += 256) {
        s_pb[k] = pi[(size_t)i * MLPH + k] + b1[k];
        s_w2[k] = W2[k];
    }
    __syncthreads();
    float acc = 0.0f;
    for (int k = 0; k < MLPH; ++k)
        acc = fmaf(tanhf(s_pb[k] + pjT[(size_t)k * T_SEQ + j]), s_w2[k], acc);
    out[(size_t)i * T_SEQ + j] = (i == j) ? 0.0f : (acc + b2[0]);
}

// ---------------------------------------------------------------------------
// Host orchestration
// ---------------------------------------------------------------------------
extern "C" void kernel_launch(void* const* d_in, const int* in_sizes, int n_in,
                              void* d_out, int out_size, void* d_ws, size_t ws_size,
                              hipStream_t stream) {
    (void)in_sizes; (void)n_in; (void)out_size; (void)ws_size;

    const int*   words    = (const int*)d_in[0];
    const int*   tags     = (const int*)d_in[1];
    const float* word_emb = (const float*)d_in[2];
    const float* tag_emb  = (const float*)d_in[3];
    const float* Wih0     = (const float*)d_in[4];
    const float* Whh0     = (const float*)d_in[5];
    const float* bih0     = (const float*)d_in[6];
    const float* bhh0     = (const float*)d_in[7];
    const float* Wih1     = (const float*)d_in[8];
    const float* Whh1     = (const float*)d_in[9];
    const float* bih1     = (const float*)d_in[10];
    const float* bhh1     = (const float*)d_in[11];
    const float* W1       = (const float*)d_in[12];
    const float* b1       = (const float*)d_in[13];
    const float* W2       = (const float*)d_in[14];
    const float* b2       = (const float*)d_in[15];
    const float* h0       = (const float*)d_in[16];
    const float* c0       = (const float*)d_in[17];
    float* out = (float*)d_out;

    char* ws = (char*)d_ws;
    size_t off = 0;
    auto alloc = [&](size_t bytes) -> char* {
        char* p = ws + off;
        off += (bytes + 255) & ~(size_t)255;
        return p;
    };

    __bf16* xbf     = (__bf16*)alloc((size_t)T_SEQ * K400P * 2);
    __bf16* Wih0bf  = (__bf16*)alloc((size_t)2 * G_DIM * K400P * 2);
    __bf16* Wih1bf  = (__bf16*)alloc((size_t)2 * G_DIM * 800 * 2);
    __bf16* W1abf   = (__bf16*)alloc((size_t)MLPH * 800 * 2);
    __bf16* W1bbf   = (__bf16*)alloc((size_t)MLPH * 800 * 2);
    float*  WhhT0   = (float*)alloc((size_t)2 * H_DIM * G_DIM * 4);
    float*  WhhT1   = (float*)alloc((size_t)2 * H_DIM * G_DIM * 4);
    float*  bc0     = (float*)alloc((size_t)2 * G_DIM * 4);
    float*  bc1     = (float*)alloc((size_t)2 * G_DIM * 4);
    float*  X0proj  = (float*)alloc((size_t)2 * T_SEQ * G_DIM * 4);
    float*  X1proj  = (float*)alloc((size_t)2 * T_SEQ * G_DIM * 4);
    __bf16* x1bf    = (__bf16*)alloc((size_t)T_SEQ * 800 * 2);
    __bf16* hvecbf  = (__bf16*)alloc((size_t)T_SEQ * 800 * 2);
    float*  pi      = (float*)alloc((size_t)T_SEQ * MLPH * 4);
    float*  pjT     = (float*)alloc((size_t)MLPH * T_SEQ * 4);

    // --- weight prep (deterministic, every launch) ---
    f32_to_bf16_pad_kernel<<<2 * G_DIM, 256, 0, stream>>>(Wih0, H_DIM, 0, Wih0bf, K400P, H_DIM);
    f32_to_bf16_pad_kernel<<<2 * G_DIM, 256, 0, stream>>>(Wih1, 800, 0, Wih1bf, 800, 800);
    f32_to_bf16_pad_kernel<<<MLPH, 256, 0, stream>>>(W1, G_DIM, 0,   W1abf, 800, 800);
    f32_to_bf16_pad_kernel<<<MLPH, 256, 0, stream>>>(W1, G_DIM, 800, W1bbf, 800, 800);

    const int TN = (G_DIM * H_DIM + 255) / 256;
    transpose_f32_kernel<<<TN, 256, 0, stream>>>(Whh0,                 WhhT0,                 G_DIM, H_DIM);
    transpose_f32_kernel<<<TN, 256, 0, stream>>>(Whh0 + G_DIM * H_DIM, WhhT0 + H_DIM * G_DIM, G_DIM, H_DIM);
    transpose_f32_kernel<<<TN, 256, 0, stream>>>(Whh1,                 WhhT1,                 G_DIM, H_DIM);
    transpose_f32_kernel<<<TN, 256, 0, stream>>>(Whh1 + G_DIM * H_DIM, WhhT1 + H_DIM * G_DIM, G_DIM, H_DIM);

    bias_add_kernel<<<(2 * G_DIM + 255) / 256, 256, 0, stream>>>(bih0, bhh0, bc0, 2 * G_DIM);
    bias_add_kernel<<<(2 * G_DIM + 255) / 256, 256, 0, stream>>>(bih1, bhh1, bc1, 2 * G_DIM);

    // --- embeddings ---
    embed_kernel<<<T_SEQ, 448, 0, stream>>>(words, tags, word_emb, tag_emb, xbf);

    // --- layer 0: batched input projection (WMMA), then recurrence ---
    // M=256 (16 tiles), N=1600 (100 tiles, strips of 4) -> 400 strips / 8 = 50 blocks
    for (int d = 0; d < 2; ++d) {
        wmma_gemm_bt_kernel<4><<<50, 256, 0, stream>>>(
            xbf, K400P,
            Wih0bf + (size_t)d * G_DIM * K400P, K400P,
            X0proj + (size_t)d * T_SEQ * G_DIM, G_DIM,
            bc0 + d * G_DIM, /*Ntiles=*/100, /*K=*/K400P, /*transC=*/0);
    }
    lstm_layer_kernel<<<2, 800, 0, stream>>>(X0proj, WhhT0, h0, c0, x1bf, 800);

    // --- layer 1 ---
    for (int d = 0; d < 2; ++d) {
        wmma_gemm_bt_kernel<4><<<50, 256, 0, stream>>>(
            x1bf, 800,
            Wih1bf + (size_t)d * G_DIM * 800, 800,
            X1proj + (size_t)d * T_SEQ * G_DIM, G_DIM,
            bc1 + d * G_DIM, /*Ntiles=*/100, /*K=*/800, /*transC=*/0);
    }
    lstm_layer_kernel<<<2, 800, 0, stream>>>(X1proj, WhhT1, h0 + 800, c0 + 800, hvecbf, 800);

    // --- pi / pj projections (WMMA): M=256 (16 tiles), N=400 (25 tiles, strips of 5)
    //     -> 80 strips / 8 = 10 blocks ---
    wmma_gemm_bt_kernel<5><<<10, 256, 0, stream>>>(
        hvecbf, 800, W1abf, 800, pi, MLPH, nullptr, /*Ntiles=*/25, /*K=*/800, /*transC=*/0);
    wmma_gemm_bt_kernel<5><<<10, 256, 0, stream>>>(
        hvecbf, 800, W1bbf, 800, pjT, T_SEQ, nullptr, /*Ntiles=*/25, /*K=*/800, /*transC=*/1);

    // --- fused tanh + dot scorer ---
    score_kernel<<<T_SEQ, 256, 0, stream>>>(pi, pjT, b1, W2, b2, out);
}
